// _ConvBlock4LessLayers_43018392436809
// MI455X (gfx1250) — matrix operand order repaired and checked by reading, not compiled, not run on hardware
//
#include <hip/hip_runtime.h>

// Problem constants from the reference (harness uses fixed setup_inputs()).
#define C64    64
#define NC     50000
#define NCP    50048   // NC rounded up to multiple of 128 (8 waves x 16 rows per block)

typedef __attribute__((ext_vector_type(16))) _Float16 v16h;
typedef __attribute__((ext_vector_type(8)))  _Float16 v8h;
typedef __attribute__((ext_vector_type(8)))  float    v8f;

// ---------------------------------------------------------------- zero fill
__global__ void zero_f32(float* __restrict__ p, long long n) {
  long long i = (long long)blockIdx.x * blockDim.x + threadIdx.x;
  long long s = (long long)gridDim.x * blockDim.x;
  for (; i < n; i += s) p[i] = 0.0f;
}

// ------------------------------------------------- weighted scatter-add (edge x channel)
// out[dst[e]][c] += w[e] * feat[src[e]][c]
__global__ void scatter_weighted(const float* __restrict__ feat,
                                 const int*   __restrict__ esrc,
                                 const int*   __restrict__ edst,
                                 const float* __restrict__ ew,
                                 float* __restrict__ out,
                                 long long nwork) {
  long long i = (long long)blockIdx.x * blockDim.x + threadIdx.x;
  long long s = (long long)gridDim.x * blockDim.x;
  for (; i < nwork; i += s) {
    int e = (int)(i >> 6);
    int c = (int)(i & 63);
    float v = ew[e] * feat[(long long)esrc[e] * C64 + c];
    unsafeAtomicAdd(&out[(long long)edst[e] * C64 + c], v);
  }
}

// ---------------------------------------------------------------- f32 -> f16 shadow
__global__ void cvt_f32_f16(const float* __restrict__ in, _Float16* __restrict__ out,
                            long long n) {
  long long i = (long long)blockIdx.x * blockDim.x + threadIdx.x;
  long long s = (long long)gridDim.x * blockDim.x;
  for (; i < n; i += s) out[i] = (_Float16)in[i];
}

// ---------------------------------------------------- pack weights into WMMA B layout
// wpack flat index: ((((layer*2 + mat)*2 + kt)*4 + nt)*32 + lane)*16 + e
// B element (k, n) of tile (kt, nt):  k = kt*32 + (lane>>4)*16 + e,  n = nt*16 + (lane&15)
__global__ void pack_weights(const float* __restrict__ Wr, const float* __restrict__ Wn,
                             _Float16* __restrict__ wpack) {
  int idx = blockIdx.x * blockDim.x + threadIdx.x;
  if (idx >= 4 * 2 * 2 * 4 * 32 * 16) return;
  int e     =  idx        & 15;
  int lane  = (idx >> 4)  & 31;
  int nt    = (idx >> 9)  & 3;
  int kt    = (idx >> 11) & 1;
  int mat   = (idx >> 12) & 1;
  int layer = (idx >> 13) & 3;
  int k = kt * 32 + (lane >> 4) * 16 + e;
  int n = nt * 16 + (lane & 15);
  const float* W = mat ? Wn : Wr;
  wpack[idx] = (_Float16)W[(layer * C64 + k) * C64 + n];
}

// -------------------------------------------------------------------- fused GEMM layer
// h_out = relu(h16 @ Wroot + agg16 @ Wnbr + bias); writes f32 master + f16 shadow.
// One wave -> 16 rows x 64 cols. wpack points at this layer's 8192 packed halves.
__global__ void gemm_relu_wmma(const _Float16* __restrict__ h16,
                               const _Float16* __restrict__ a16,
                               const _Float16* __restrict__ wpack,
                               const float*    __restrict__ bias,
                               float*          __restrict__ hout,
                               _Float16*       __restrict__ hout16,
                               int nrows) {
  int gwave = (int)(((long long)blockIdx.x * blockDim.x + threadIdx.x) >> 5);
  int lane  = threadIdx.x & 31;
  int m0 = gwave * 16;
  if (m0 >= nrows) return;                  // uniform per-wave exit, EXEC stays all-1s
  int mr = lane & 15;
  int hs = lane >> 4;

  const _Float16* hrow = h16 + (size_t)(m0 + mr) * C64;
  const _Float16* arow = a16 + (size_t)(m0 + mr) * C64;

  v8f acc[4];
#pragma unroll
  for (int nt = 0; nt < 4; ++nt)
#pragma unroll
    for (int r = 0; r < 8; ++r) acc[nt][r] = 0.0f;

#pragma unroll
  for (int kt = 0; kt < 2; ++kt) {
    const int k0 = kt * 32;
    // A operands: lane-group hs holds contiguous K = hs*16 + e (matches packed B).
    v16h Ah, Aa;
    {
      v8h lo = *(const v8h*)(hrow + k0 + hs * 16);
      v8h hi = *(const v8h*)(hrow + k0 + hs * 16 + 8);
#pragma unroll
      for (int i = 0; i < 8; ++i) { Ah[i] = lo[i]; Ah[8 + i] = hi[i]; }
      lo = *(const v8h*)(arow + k0 + hs * 16);
      hi = *(const v8h*)(arow + k0 + hs * 16 + 8);
#pragma unroll
      for (int i = 0; i < 8; ++i) { Aa[i] = lo[i]; Aa[8 + i] = hi[i]; }
    }
#pragma unroll
    for (int nt = 0; nt < 4; ++nt) {
      const _Float16* br = wpack + (size_t)((((0 * 2 + kt) * 4 + nt) * 32 + lane) * 16);
      const _Float16* bn = wpack + (size_t)((((1 * 2 + kt) * 4 + nt) * 32 + lane) * 16);
      v16h Br, Bn;
      {
        v8h l0 = *(const v8h*)br, h0 = *(const v8h*)(br + 8);
#pragma unroll
        for (int i = 0; i < 8; ++i) { Br[i] = l0[i]; Br[8 + i] = h0[i]; }
        l0 = *(const v8h*)bn; h0 = *(const v8h*)(bn + 8);
#pragma unroll
        for (int i = 0; i < 8; ++i) { Bn[i] = l0[i]; Bn[8 + i] = h0[i]; }
      }
      acc[nt] = __builtin_amdgcn_wmma_f32_16x16x32_f16(false, Ah, false, Br,
                                                       (short)0, acc[nt], false, false);
      acc[nt] = __builtin_amdgcn_wmma_f32_16x16x32_f16(false, Aa, false, Bn,
                                                       (short)0, acc[nt], false, false);
    }
  }

  // Epilogue: C/D layout  M = hs*8 + r, N = nt*16 + (lane&15)
#pragma unroll
  for (int nt = 0; nt < 4; ++nt) {
    int n = nt * 16 + (lane & 15);
    float bv = bias[n];
#pragma unroll
    for (int r = 0; r < 8; ++r) {
      int row = m0 + hs * 8 + r;
      float v = acc[nt][r] + bv;
      v = v > 0.0f ? v : 0.0f;
      hout  [(size_t)row * C64 + n] = v;
      hout16[(size_t)row * C64 + n] = (_Float16)v;
    }
  }
}

// ===================================================================== launcher
extern "C" void kernel_launch(void* const* d_in, const int* in_sizes, int n_in,
                              void* d_out, int out_size, void* d_ws, size_t ws_size,
                              hipStream_t stream) {
  const float* x          = (const float*)d_in[0];
  const int*   pool_src   = (const int*)  d_in[1];
  const int*   pool_dst   = (const int*)  d_in[2];
  const float* pool_w     = (const float*)d_in[3];
  const int*   pp_src     = (const int*)  d_in[4];
  const int*   pp_dst     = (const int*)  d_in[5];
  const float* pp_w       = (const float*)d_in[6];
  const int*   unpool_src = (const int*)  d_in[7];
  const int*   unpool_dst = (const int*)  d_in[8];
  const float* unpool_w   = (const float*)d_in[9];
  const float* W_root     = (const float*)d_in[10];
  const float* W_nbr      = (const float*)d_in[11];
  const float* bias       = (const float*)d_in[12];
  float* out = (float*)d_out;

  const long long E_pool   = in_sizes[1];
  const long long E_pp     = in_sizes[4];
  const long long E_unpool = in_sizes[7];
  const long long n_fine_elems = in_sizes[0];        // N_F * 64

  // -------- workspace carve-up (all 256B aligned)
  char* ws = (char*)d_ws;
  size_t off = 0;
  auto carve = [&](size_t bytes) { char* p = ws + off; off = (off + bytes + 255) & ~(size_t)255; return p; };
  float*    h0    = (float*)   carve((size_t)NCP * C64 * 4);
  float*    h1    = (float*)   carve((size_t)NCP * C64 * 4);
  float*    agg   = (float*)   carve((size_t)NCP * C64 * 4);
  _Float16* h16a  = (_Float16*)carve((size_t)NCP * C64 * 2);
  _Float16* h16b  = (_Float16*)carve((size_t)NCP * C64 * 2);
  _Float16* agg16 = (_Float16*)carve((size_t)NCP * C64 * 2);
  _Float16* wpack = (_Float16*)carve((size_t)4 * 2 * 2 * 4 * 32 * 16 * 2);
  (void)ws_size;

  const int TPB = 256;
  const long long nCoarseElems = (long long)NCP * C64;
  const int zBlocksC = (int)((nCoarseElems + TPB - 1) / TPB);

  // 1) pool: h0 = scatter(x)
  zero_f32<<<zBlocksC, TPB, 0, stream>>>(h0, nCoarseElems);
  {
    long long nw = E_pool * C64;
    scatter_weighted<<<(int)((nw + TPB - 1) / TPB), TPB, 0, stream>>>(
        x, pool_src, pool_dst, pool_w, h0, nw);
  }
  cvt_f32_f16<<<zBlocksC, TPB, 0, stream>>>(h0, h16a, nCoarseElems);

  // 2) pack all layer weights into WMMA B layout
  pack_weights<<<(4 * 2 * 2 * 4 * 32 * 16) / TPB, TPB, 0, stream>>>(W_root, W_nbr, wpack);

  // 3) four graph-conv layers
  float*    hc  = h0;   float*    hn  = h1;
  _Float16* hc16 = h16a; _Float16* hn16 = h16b;
  const int gemmWaves  = NCP / 16;
  const int gemmBlocks = (gemmWaves * 32 + TPB - 1) / TPB;
  for (int layer = 0; layer < 4; ++layer) {
    zero_f32<<<zBlocksC, TPB, 0, stream>>>(agg, nCoarseElems);
    {
      long long nw = E_pp * C64;
      scatter_weighted<<<(int)((nw + TPB - 1) / TPB), TPB, 0, stream>>>(
          hc, pp_src, pp_dst, pp_w, agg, nw);
    }
    cvt_f32_f16<<<zBlocksC, TPB, 0, stream>>>(agg, agg16, nCoarseElems);
    gemm_relu_wmma<<<gemmBlocks, TPB, 0, stream>>>(
        hc16, agg16, wpack + (size_t)layer * 2 * 2 * 4 * 32 * 16,
        bias + (size_t)layer * C64, hn, hn16, NCP);
    { float* t = hc; hc = hn; hn = t; }
    { _Float16* t = hc16; hc16 = hn16; hn16 = t; }
  }

  // 4) unpool: out = scatter(h_final)
  zero_f32<<<(int)((n_fine_elems + TPB - 1) / TPB), TPB, 0, stream>>>(out, n_fine_elems);
  {
    long long nw = E_unpool * C64;
    scatter_weighted<<<(int)((nw + TPB - 1) / TPB), TPB, 0, stream>>>(
        hc, unpool_src, unpool_dst, unpool_w, out, nw);
  }
}
